// Attention_67327907332706
// MI455X (gfx1250) — matrix operand order repaired and checked by reading, not compiled, not run on hardware
//
#include <hip/hip_runtime.h>
#include <hip/hip_bf16.h>

#define DIMC 768
#define NH 12
#define HD 64
#define SEQ 4096
#define BATCH 2
// softmax computed in base-2: score * (1/sqrt(64)) * log2(e)
#define KSCALE 0.18033688011112042f

typedef __attribute__((ext_vector_type(16))) __bf16 v16bf;
typedef __attribute__((ext_vector_type(8)))  __bf16 v8bf;
typedef __attribute__((ext_vector_type(4)))  __bf16 v4bf;
typedef __attribute__((ext_vector_type(8)))  float  v8f;
typedef __attribute__((ext_vector_type(4)))  float  v4f;

static __device__ inline v8f wmma_bf16(v16bf a, v16bf b, v8f c) {
  return __builtin_amdgcn_wmma_f32_16x16x32_bf16(
      false, a, false, b, (short)0, c, false, false);
}

static __device__ inline v8f vzero8() {
  v8f v;
#pragma unroll
  for (int i = 0; i < 8; ++i) v[i] = 0.0f;
  return v;
}

// Async global->LDS copy of 16 bytes per lane (CDNA5, ASYNCcnt-tracked).
static __device__ inline void async_copy_b128(unsigned lds_byte_addr,
                                              const void* gaddr) {
  asm volatile("global_load_async_to_lds_b128 %0, %1, off"
               :: "v"(lds_byte_addr), "v"((unsigned long long)(uintptr_t)gaddr)
               : "memory");
}
template <int N> static __device__ inline void wait_async() {
  asm volatile("s_wait_asynccnt %0" :: "i"(N) : "memory");
}

// A-fragment (16x32 bf16): lane holds row = lane%16, g = lane/16.
// elems 0..7 -> K = 8g+i ; elems 8..15 -> K = 16+8g+(i-8). Two b128 loads.
static __device__ inline v16bf load_afrag(const __bf16* row_base, int g) {
  v8bf lo = *(const v8bf*)(row_base + 8 * g);
  v8bf hi = *(const v8bf*)(row_base + 8 * g + 16);
  v16bf a;
#pragma unroll
  for (int i = 0; i < 8; ++i) { a[i] = lo[i]; a[8 + i] = hi[i]; }
  return a;
}

// B-fragment (32x16 bf16) from Bt layout (Bt[n][k]): lane holds col = lane%16,
// elems 0..7 -> K = 8g+i ; elems 8..15 -> K = 16+8g+(i-8). Two b128 loads.
static __device__ inline v16bf load_bfrag(const __bf16* col_base, int g) {
  v8bf lo = *(const v8bf*)(col_base + 8 * g);
  v8bf hi = *(const v8bf*)(col_base + 8 * g + 16);
  v16bf b;
#pragma unroll
  for (int i = 0; i < 8; ++i) { b[i] = lo[i]; b[8 + i] = hi[i]; }
  return b;
}

// ---------------------------------------------------------------------------
// Kernel 1: QKV projection. x(fp32 8192x768) @ Wqkv(fp32 768x2304) -> bf16
// Q/K/V in [B,H,N,D]. Block tile 128x128, 8 waves of 32x64.
// Double-buffered LDS + register prefetch of the next k-tile.
// ---------------------------------------------------------------------------
__global__ __launch_bounds__(256) void qkv_gemm_kernel(
    const float* __restrict__ x, const float* __restrict__ Wqkv,
    __bf16* __restrict__ Qo, __bf16* __restrict__ Ko, __bf16* __restrict__ Vo) {
  __shared__ __align__(16) __bf16 As[2][128][40];
  __shared__ __align__(16) __bf16 Bs[2][128][40];   // transposed: Bs[n][k]
  const int tid = threadIdx.x;
  const int wave = tid >> 5, lane = tid & 31;
  const int g = lane >> 4, c = lane & 15;
  const int m0 = blockIdx.x * 128;
  const int n0 = blockIdx.y * 128;
  const int wm = (wave >> 1) * 32;
  const int wn = (wave & 1) * 64;
  // staging coordinates (chunk i adds 32 rows / 8 k-rows)
  const int arow = tid >> 3, acol = (tid & 7) * 4;
  const int brow = tid >> 5, bcol = (tid & 31) * 4;

  v8f acc[2][4];
#pragma unroll
  for (int i = 0; i < 2; ++i)
#pragma unroll
    for (int j = 0; j < 4; ++j) acc[i][j] = vzero8();

  v4f ra[4], rb[4];
#pragma unroll
  for (int i = 0; i < 4; ++i) {
    ra[i] = *(const v4f*)&x[(size_t)(m0 + arow + 32 * i) * DIMC + acol];
    rb[i] = *(const v4f*)&Wqkv[(size_t)(brow + 8 * i) * (3 * DIMC) + n0 + bcol];
  }

  int buf = 0;
  for (int kt = 0; kt < DIMC / 32; ++kt) {
    // commit prefetched tile kt to LDS[buf]
#pragma unroll
    for (int i = 0; i < 4; ++i) {
      v4bf h;
#pragma unroll
      for (int j = 0; j < 4; ++j) h[j] = (__bf16)ra[i][j];
      *(v4bf*)&As[buf][arow + 32 * i][acol] = h;
    }
#pragma unroll
    for (int i = 0; i < 4; ++i)
#pragma unroll
      for (int j = 0; j < 4; ++j)
        Bs[buf][bcol + j][brow + 8 * i] = (__bf16)rb[i][j];
    // prefetch tile kt+1 into registers (in flight during WMMAs)
    if (kt + 1 < DIMC / 32) {
      int k0 = (kt + 1) * 32;
#pragma unroll
      for (int i = 0; i < 4; ++i) {
        ra[i] = *(const v4f*)&x[(size_t)(m0 + arow + 32 * i) * DIMC + k0 + acol];
        rb[i] = *(const v4f*)&Wqkv[(size_t)(k0 + brow + 8 * i) * (3 * DIMC) + n0 + bcol];
      }
    }
    __syncthreads();
    v16bf a0 = load_afrag(&As[buf][wm + c][0], g);
    v16bf a1 = load_afrag(&As[buf][wm + 16 + c][0], g);
#pragma unroll
    for (int ct = 0; ct < 4; ++ct) {
      v16bf bfr = load_bfrag(&Bs[buf][wn + ct * 16 + c][0], g);
      acc[0][ct] = wmma_bf16(a0, bfr, acc[0][ct]);
      acc[1][ct] = wmma_bf16(a1, bfr, acc[1][ct]);
    }
    buf ^= 1;
  }

#pragma unroll
  for (int rh = 0; rh < 2; ++rh)
#pragma unroll
    for (int ct = 0; ct < 4; ++ct) {
      int col = n0 + wn + ct * 16 + c;          // 0..2303
      int three = col / DIMC;
      int rem = col - three * DIMC;
      int h = rem >> 6, d = rem & 63;
      __bf16* dst = (three == 0) ? Qo : (three == 1) ? Ko : Vo;
#pragma unroll
      for (int v = 0; v < 8; ++v) {
        int r = m0 + wm + rh * 16 + v + 8 * g;
        int b = r >> 12;
        int n = r & (SEQ - 1);
        dst[(((size_t)(b * NH + h)) * SEQ + n) * HD + d] = (__bf16)acc[rh][ct][v];
      }
    }
}

// ---------------------------------------------------------------------------
// Kernel 2: flash attention. One block = 128 queries of one (b,h).
// 8 waves x 16 queries; keys streamed in DOUBLE-BUFFERED tiles of 64:
// K tile by async global->LDS DMA, V tile transposed via b128 loads.
// base-2 online softmax.
// ---------------------------------------------------------------------------
__global__ __launch_bounds__(256) void flash_attn_kernel(
    const __bf16* __restrict__ Qg, const __bf16* __restrict__ Kg,
    const __bf16* __restrict__ Vg, __bf16* __restrict__ Tok) {
  __shared__ __align__(16) __bf16 Ks[2][64][72];    // [key][d]  == Bt for Q*K^T
  __shared__ __align__(16) __bf16 Vts[2][64][72];   // [d][key]  == Bt for P*V
  __shared__ __align__(16) __bf16 Ps[8][16][72];    // per-wave P scratch
  const int tid = threadIdx.x;
  const int wave = tid >> 5, lane = tid & 31;
  const int g = lane >> 4, c = lane & 15;
  const int bh = blockIdx.y;
  const int b = bh / NH, h = bh - b * NH;
  const size_t base = (size_t)bh * SEQ * HD;
  const __bf16* Qb = Qg + base;
  const __bf16* Kb = Kg + base;
  const __bf16* Vb = Vg + base;
  const int q0 = blockIdx.x * 128 + wave * 16;

  // staging coordinates: 16B chunk; chunk p adds 32 key-rows
  const int skk = tid >> 3;          // key row 0..31 (+32 for p=1)
  const int sdd = (tid & 7) * 8;     // d offset 0,8,..,56

  // resident Q fragments: 16 rows x 64 d = 2 A-fragments
  v16bf qf[2];
#pragma unroll
  for (int ch = 0; ch < 2; ++ch)
    qf[ch] = load_afrag(Qb + (size_t)(q0 + c) * HD + ch * 32, g);

  v8f o[4];
#pragma unroll
  for (int dt = 0; dt < 4; ++dt) o[dt] = vzero8();
  float mst[8], lst[8];
#pragma unroll
  for (int v = 0; v < 8; ++v) { mst[v] = -__builtin_inff(); lst[v] = 0.0f; }

  // prologue: stage tile 0 into buffer 0
#pragma unroll
  for (int p = 0; p < 2; ++p) {
    int kk = skk + 32 * p;
    async_copy_b128((unsigned)(uintptr_t)&Ks[0][kk][sdd],
                    Kb + (size_t)kk * HD + sdd);
    v8bf vv = *(const v8bf*)(Vb + (size_t)kk * HD + sdd);
#pragma unroll
    for (int j = 0; j < 8; ++j) Vts[0][sdd + j][kk] = vv[j];
  }
  wait_async<0>();
  __syncthreads();

  const int NT = SEQ / 64;
  for (int kt = 0; kt < NT; ++kt) {
    const int cur = kt & 1;
    // stage tile kt+1 into the other buffer (overlaps with compute)
    if (kt + 1 < NT) {
      int key0 = (kt + 1) * 64;
#pragma unroll
      for (int p = 0; p < 2; ++p) {
        int kk = skk + 32 * p;
        async_copy_b128((unsigned)(uintptr_t)&Ks[cur ^ 1][kk][sdd],
                        Kb + (size_t)(key0 + kk) * HD + sdd);
        v8bf vv = *(const v8bf*)(Vb + (size_t)(key0 + kk) * HD + sdd);
#pragma unroll
        for (int j = 0; j < 8; ++j) Vts[cur ^ 1][sdd + j][kk] = vv[j];
      }
    }

    // S = Q * K^T  (16 queries x 64 keys), f32 accum
    v8f s[4] = { vzero8(), vzero8(), vzero8(), vzero8() };
#pragma unroll
    for (int ct = 0; ct < 4; ++ct)
#pragma unroll
      for (int ch = 0; ch < 2; ++ch) {
        v16bf bfr = load_bfrag(&Ks[cur][ct * 16 + c][ch * 32], g);
        s[ct] = wmma_bf16(qf[ch], bfr, s[ct]);
      }

    // base-2 online softmax: row (v + 8g) spans 16 lanes of the half-wave
#pragma unroll
    for (int v = 0; v < 8; ++v) {
      float sc0 = s[0][v] * KSCALE;
      float sc1 = s[1][v] * KSCALE;
      float sc2 = s[2][v] * KSCALE;
      float sc3 = s[3][v] * KSCALE;
      float rm = fmaxf(fmaxf(sc0, sc1), fmaxf(sc2, sc3));
#pragma unroll
      for (int msk = 8; msk >= 1; msk >>= 1)
        rm = fmaxf(rm, __shfl_xor(rm, msk, 16));
      float mnew = fmaxf(mst[v], rm);
      float alpha = __builtin_amdgcn_exp2f(mst[v] - mnew);
      mst[v] = mnew;
      float p0 = __builtin_amdgcn_exp2f(sc0 - mnew);
      float p1 = __builtin_amdgcn_exp2f(sc1 - mnew);
      float p2 = __builtin_amdgcn_exp2f(sc2 - mnew);
      float p3 = __builtin_amdgcn_exp2f(sc3 - mnew);
      float rs = (p0 + p1) + (p2 + p3);
#pragma unroll
      for (int msk = 8; msk >= 1; msk >>= 1)
        rs += __shfl_xor(rs, msk, 16);
      lst[v] = lst[v] * alpha + rs;
#pragma unroll
      for (int dt = 0; dt < 4; ++dt) o[dt][v] *= alpha;
      // C-layout -> row-major LDS (wave-private; per-wave LDS ops are in-order)
      Ps[wave][v + 8 * g][c]      = (__bf16)p0;
      Ps[wave][v + 8 * g][16 + c] = (__bf16)p1;
      Ps[wave][v + 8 * g][32 + c] = (__bf16)p2;
      Ps[wave][v + 8 * g][48 + c] = (__bf16)p3;
    }

    // O += P * V   (16 x 64) x (64 keys x 64 d); Ps is wave-private so no
    // block barrier is needed between the stores above and these loads.
    v16bf pa0 = load_afrag(&Ps[wave][c][0], g);
    v16bf pa1 = load_afrag(&Ps[wave][c][32], g);
#pragma unroll
    for (int dt = 0; dt < 4; ++dt) {
      v16bf b0 = load_bfrag(&Vts[cur][dt * 16 + c][0], g);
      v16bf b1 = load_bfrag(&Vts[cur][dt * 16 + c][32], g);
      o[dt] = wmma_bf16(pa0, b0, o[dt]);
      o[dt] = wmma_bf16(pa1, b1, o[dt]);
    }

    wait_async<0>();   // next K tile landed in LDS
    __syncthreads();   // all waves done with cur; next buffer ready
  }

  // finalize: O /= l, write tokens [B,N,C] with merged heads
#pragma unroll
  for (int dt = 0; dt < 4; ++dt) {
    int col = h * HD + dt * 16 + c;
#pragma unroll
    for (int v = 0; v < 8; ++v) {
      int row = q0 + v + 8 * g;
      float val = o[dt][v] / lst[v];
      Tok[((size_t)(b * SEQ + row)) * DIMC + col] = (__bf16)val;
    }
  }
}

// ---------------------------------------------------------------------------
// Kernel 3: output projection. tokens(bf16 8192x768) @ Wp(768x768) + bias -> f32
// Block tile 128x128; A tile async-DMA'd into double-buffered LDS,
// B tile register-prefetched.
// ---------------------------------------------------------------------------
__global__ __launch_bounds__(256) void proj_gemm_kernel(
    const __bf16* __restrict__ Tok, const float* __restrict__ Wp,
    const float* __restrict__ bias, float* __restrict__ out) {
  __shared__ __align__(16) __bf16 As[2][128][40];
  __shared__ __align__(16) __bf16 Bs[2][128][40];
  const int tid = threadIdx.x;
  const int wave = tid >> 5, lane = tid & 31;
  const int g = lane >> 4, c = lane & 15;
  const int m0 = blockIdx.x * 128;
  const int n0 = blockIdx.y * 128;
  const int wm = (wave >> 1) * 32;
  const int wn = (wave & 1) * 64;
  const int arow = tid >> 2, add8 = (tid & 3) * 8;   // A: 2 chunks (+64 rows)
  const int brow = tid >> 5, bcol = (tid & 31) * 4;  // B: 4 chunks (+8 k-rows)

  v8f acc[2][4];
#pragma unroll
  for (int i = 0; i < 2; ++i)
#pragma unroll
    for (int j = 0; j < 4; ++j) acc[i][j] = vzero8();

  // prologue: async A tile 0 -> As[0]; prefetch B tile 0 into regs
#pragma unroll
  for (int i = 0; i < 2; ++i)
    async_copy_b128((unsigned)(uintptr_t)&As[0][arow + 64 * i][add8],
                    Tok + (size_t)(m0 + arow + 64 * i) * DIMC + add8);
  v4f rb[4];
#pragma unroll
  for (int i = 0; i < 4; ++i)
    rb[i] = *(const v4f*)&Wp[(size_t)(brow + 8 * i) * DIMC + n0 + bcol];

  int buf = 0;
  for (int kt = 0; kt < DIMC / 32; ++kt) {
    // commit prefetched B tile to LDS[buf]
#pragma unroll
    for (int i = 0; i < 4; ++i)
#pragma unroll
      for (int j = 0; j < 4; ++j)
        Bs[buf][bcol + j][brow + 8 * i] = (__bf16)rb[i][j];
    // prefetch tile kt+1 (async A into other buffer; B into regs)
    if (kt + 1 < DIMC / 32) {
      int k0 = (kt + 1) * 32;
#pragma unroll
      for (int i = 0; i < 2; ++i)
        async_copy_b128((unsigned)(uintptr_t)&As[buf ^ 1][arow + 64 * i][add8],
                        Tok + (size_t)(m0 + arow + 64 * i) * DIMC + k0 + add8);
#pragma unroll
      for (int i = 0; i < 4; ++i)
        rb[i] = *(const v4f*)&Wp[(size_t)(k0 + brow + 8 * i) * DIMC + n0 + bcol];
      wait_async<2>();   // retire tile kt's 2 async copies (in-order)
    } else {
      wait_async<0>();
    }
    __syncthreads();
    v16bf a0 = load_afrag(&As[buf][wm + c][0], g);
    v16bf a1 = load_afrag(&As[buf][wm + 16 + c][0], g);
#pragma unroll
    for (int ct = 0; ct < 4; ++ct) {
      v16bf bfr = load_bfrag(&Bs[buf][wn + ct * 16 + c][0], g);
      acc[0][ct] = wmma_bf16(a0, bfr, acc[0][ct]);
      acc[1][ct] = wmma_bf16(a1, bfr, acc[1][ct]);
    }
    buf ^= 1;
  }

#pragma unroll
  for (int rh = 0; rh < 2; ++rh)
#pragma unroll
    for (int ct = 0; ct < 4; ++ct) {
      int col = n0 + wn + ct * 16 + c;
      float bb = bias[col];
#pragma unroll
      for (int v = 0; v < 8; ++v) {
        int r = m0 + wm + rh * 16 + v + 8 * g;
        out[(size_t)r * DIMC + col] = acc[rh][ct][v] + bb;
      }
    }
}

// ---------------------------------------------------------------------------
extern "C" void kernel_launch(void* const* d_in, const int* in_sizes, int n_in,
                              void* d_out, int out_size, void* d_ws, size_t ws_size,
                              hipStream_t stream) {
  (void)in_sizes; (void)n_in; (void)out_size; (void)ws_size;
  const float* x    = (const float*)d_in[0];
  const float* Wqkv = (const float*)d_in[1];
  const float* Wp   = (const float*)d_in[2];
  const float* bp   = (const float*)d_in[3];
  float* out = (float*)d_out;

  const size_t nTok = (size_t)BATCH * SEQ * DIMC;  // 6,291,456 elems
  __bf16* Qw  = (__bf16*)d_ws;        // [B,H,N,D] bf16
  __bf16* Kw  = Qw + nTok;
  __bf16* Vw  = Kw + nTok;
  __bf16* Tok = Vw + nTok;            // [B,N,C] bf16

  qkv_gemm_kernel<<<dim3((BATCH * SEQ) / 128, (3 * DIMC) / 128), 256, 0, stream>>>(
      x, Wqkv, Qw, Kw, Vw);
  flash_attn_kernel<<<dim3(SEQ / 128, BATCH * NH), 256, 0, stream>>>(
      Qw, Kw, Vw, Tok);
  proj_gemm_kernel<<<dim3((BATCH * SEQ) / 128, DIMC / 128), 256, 0, stream>>>(
      Tok, Wp, bp, out);
}
